// GAT_Net_17617955848510
// MI455X (gfx1250) — compile-verified, matrix operand order checked by reading, not run on hardware
//
#include <hip/hip_runtime.h>
#include <hip/hip_bf16.h>
#include <stdint.h>

#define NNODE 100000
#define NEDGE 1600000
#define E2    (NEDGE + NNODE)   // with self loops
#define FIN   128
#define HIDD  64
#define HEADS 2
#define NCLS  18
#define NEG_SLOPE 0.2f

typedef __attribute__((ext_vector_type(16))) __bf16 v16bf;
typedef __attribute__((ext_vector_type(8)))  float  v8f;

union FragBF { v16bf v; uint4 u[2]; };

// ---- monotonic float <-> uint encoding so atomicMax(uint) == float max ----
// encode(-inf)=0x007FFFFF > 0, and 0 (memset) is below every real encoding.
__device__ __forceinline__ unsigned enc_ordered(float f) {
  int b = __float_as_int(f);
  unsigned ub = (unsigned)b;
  return (b >= 0) ? (ub | 0x80000000u) : ~ub;
}
__device__ __forceinline__ float dec_ordered(unsigned u) {
  unsigned b = (u & 0x80000000u) ? (u & 0x7FFFFFFFu) : ~u;
  return __uint_as_float(b);
}

__device__ __forceinline__ void edge_sd(const long long* __restrict__ ei, long e, int& s, int& d) {
  if (e < (long)NEDGE) { s = (int)ei[e]; d = (int)ei[(long)NEDGE + e]; }
  else                 { s = d = (int)(e - NEDGE); }
}

// ---------------- prep kernels ----------------
__global__ void cvt_bf16_kernel(const float* __restrict__ s, __bf16* __restrict__ d, long n) {
  long i = (long)blockIdx.x * blockDim.x + threadIdx.x;
  if (i < n) d[i] = (__bf16)s[i];
}

// B is [K][C] f32 row-major; Bt is [Cpad][K] bf16 row-major (rows >= C zero-filled)
__global__ void transpose_bf16_kernel(const float* __restrict__ B, __bf16* __restrict__ Bt,
                                      int K, int C, int Cpad) {
  int i = blockIdx.x * blockDim.x + threadIdx.x;
  if (i >= Cpad * K) return;
  int c = i / K, k = i - c * K;
  float v = (c < C) ? B[(long)k * C + c] : 0.0f;
  Bt[(long)c * K + k] = (__bf16)v;
}

// ---------------- WMMA GEMM ----------------
// C[M x NC] = A[M x K](bf16,row-major) * Bt[NC x K](bf16,row-major)^T
// blockDim.x = 32 * (NC/16); one wave per 16x16 tile; M % 16 == 0, K % 32 == 0.
// mode 0: store f32 to Cf, leading dim NStore, cols >= NStore masked off.
// mode 1: add col-bias, relu, store bf16 to Cb, leading dim NC.
__global__ __launch_bounds__(256) void wmma_gemm_bf16_kernel(
    const __bf16* __restrict__ A, const __bf16* __restrict__ Bt,
    float* __restrict__ Cf, __bf16* __restrict__ Cb,
    const float* __restrict__ bias,
    int K, int NC, int NStore, int mode) {
  const int lane    = threadIdx.x & 31;
  const int wave    = threadIdx.x >> 5;
  const int rowbase = blockIdx.x << 4;
  const int colbase = wave << 4;
  const int m   = rowbase + (lane & 15);
  const int n   = colbase + (lane & 15);
  const int klo = (lane & 16) ? 8 : 0;   // per-lane K sub-offset (16-bit A/B layout)

  const uint4* __restrict__ Arow = (const uint4*)(A + (size_t)m * K);
  const uint4* __restrict__ Brow = (const uint4*)(Bt + (size_t)n * K);

  v8f acc = {};
  for (int kb = 0; kb < K; kb += 32) {
    int i0 = (kb + klo) >> 3;            // uint4 index: 8 halfs each
    FragBF a, b;
    a.u[0] = Arow[i0]; a.u[1] = Arow[i0 + 2];   // K = base+0..7 and base+16..23
    b.u[0] = Brow[i0]; b.u[1] = Brow[i0 + 2];
    acc = __builtin_amdgcn_wmma_f32_16x16x32_bf16(false, a.v, false, b.v,
                                                  (short)0, acc, false, false);
  }

  const int r0 = (lane & 16) ? 8 : 0;    // C/D layout: elem r -> row r (+8 for hi lanes)
  if (mode == 0) {
    if (n < NStore) {
      #pragma unroll
      for (int r = 0; r < 8; ++r)
        Cf[(size_t)(rowbase + r0 + r) * NStore + n] = acc[r];
    }
  } else {
    float bv = bias ? bias[n] : 0.0f;
    #pragma unroll
    for (int r = 0; r < 8; ++r)
      Cb[(size_t)(rowbase + r0 + r) * NC + n] = (__bf16)fmaxf(acc[r] + bv, 0.0f);
  }
}

// ---------------- attention scores ----------------
__global__ void attn_scores1_kernel(const float* __restrict__ h1,
                                    const float* __restrict__ att_s,
                                    const float* __restrict__ att_d,
                                    float* __restrict__ as1, float* __restrict__ ad1) {
  int i = blockIdx.x * blockDim.x + threadIdx.x;   // N*HEADS
  if (i >= NNODE * HEADS) return;
  int node = i >> 1, head = i & 1;
  const float* hp = h1 + (size_t)node * (HEADS * HIDD) + head * HIDD;
  const float* sp = att_s + head * HIDD;
  const float* dp = att_d + head * HIDD;
  float s = 0.f, d = 0.f;
  #pragma unroll 4
  for (int c = 0; c < HIDD; ++c) { float v = hp[c]; s += v * sp[c]; d += v * dp[c]; }
  as1[i] = s; ad1[i] = d;
}

__global__ void attn_scores2_kernel(const float* __restrict__ h2,
                                    const float* __restrict__ att_s,
                                    const float* __restrict__ att_d,
                                    float* __restrict__ as2, float* __restrict__ ad2) {
  int n = blockIdx.x * blockDim.x + threadIdx.x;
  if (n >= NNODE) return;
  const float* hp = h2 + (size_t)n * NCLS;
  float s = 0.f, d = 0.f;
  #pragma unroll
  for (int c = 0; c < NCLS; ++c) { float v = hp[c]; s += v * att_s[c]; d += v * att_d[c]; }
  as2[n] = s; ad2[n] = d;
}

// ---------------- layer-1 edge phase ----------------
__global__ void edge_max1_kernel(const long long* __restrict__ ei,
                                 const float* __restrict__ as1, const float* __restrict__ ad1,
                                 unsigned* __restrict__ m1) {
  long i = (long)blockIdx.x * blockDim.x + threadIdx.x;    // E2*HEADS
  if (i >= (long)E2 * HEADS) return;
  long e = i >> 1; int h = (int)(i & 1);
  int s, d; edge_sd(ei, e, s, d);
  float v = as1[s * 2 + h] + ad1[d * 2 + h];
  v = (v > 0.f) ? v : NEG_SLOPE * v;
  atomicMax(&m1[d * 2 + h], enc_ordered(v));
}

__global__ void edge_exp1_kernel(const long long* __restrict__ ei,
                                 const float* __restrict__ as1, const float* __restrict__ ad1,
                                 const unsigned* __restrict__ m1,
                                 float* __restrict__ ex1, float* __restrict__ denom1) {
  long i = (long)blockIdx.x * blockDim.x + threadIdx.x;
  if (i >= (long)E2 * HEADS) return;
  long e = i >> 1; int h = (int)(i & 1);
  int s, d; edge_sd(ei, e, s, d);
  float v = as1[s * 2 + h] + ad1[d * 2 + h];
  v = (v > 0.f) ? v : NEG_SLOPE * v;
  float ex = expf(v - dec_ordered(m1[d * 2 + h]));
  ex1[i] = ex;
  atomicAdd(&denom1[d * 2 + h], ex);
}

// one wave per edge; lane handles 4 of the 128 channels (float4 gather + 4 atomics)
__global__ void edge_agg1_kernel(const long long* __restrict__ ei,
                                 const float* __restrict__ h1,
                                 const float* __restrict__ ex1,
                                 const float* __restrict__ denom1,
                                 float* __restrict__ agg1) {
  long g = (long)blockIdx.x * blockDim.x + threadIdx.x;
  long e = g >> 5; int lane = (int)(g & 31);
  if (e >= (long)E2) return;
  int s, d; edge_sd(ei, e, s, d);
  int c0 = lane * 4;
  int h  = c0 >> 6;                        // head 0: c 0..63, head 1: c 64..127
  float coef = ex1[e * 2 + h] / denom1[d * 2 + h];
  const float4 hv = *(const float4*)(h1 + (size_t)s * (HEADS * HIDD) + c0);
  float* ap = agg1 + (size_t)d * (HEADS * HIDD) + c0;
  atomicAdd(ap + 0, hv.x * coef);
  atomicAdd(ap + 1, hv.y * coef);
  atomicAdd(ap + 2, hv.z * coef);
  atomicAdd(ap + 3, hv.w * coef);
}

// bias + ELU, emit bf16 for next GEMM
__global__ void node_elu1_kernel(const float* __restrict__ agg1, const float* __restrict__ b1,
                                 __bf16* __restrict__ h1ab) {
  long i = (long)blockIdx.x * blockDim.x + threadIdx.x;   // N*128
  if (i >= (long)NNODE * (HEADS * HIDD)) return;
  int c = (int)(i & (HEADS * HIDD - 1));
  float v = agg1[i] + b1[c];
  v = (v > 0.f) ? v : (expf(v) - 1.0f);
  h1ab[i] = (__bf16)v;
}

// ---------------- layer-2 edge phase ----------------
__global__ void edge_max2_kernel(const long long* __restrict__ ei,
                                 const float* __restrict__ as2, const float* __restrict__ ad2,
                                 unsigned* __restrict__ m2) {
  long e = (long)blockIdx.x * blockDim.x + threadIdx.x;
  if (e >= (long)E2) return;
  int s, d; edge_sd(ei, e, s, d);
  float v = as2[s] + ad2[d];
  v = (v > 0.f) ? v : NEG_SLOPE * v;
  atomicMax(&m2[d], enc_ordered(v));
}

__global__ void edge_exp2_kernel(const long long* __restrict__ ei,
                                 const float* __restrict__ as2, const float* __restrict__ ad2,
                                 const unsigned* __restrict__ m2,
                                 float* __restrict__ ex2, float* __restrict__ denom2) {
  long e = (long)blockIdx.x * blockDim.x + threadIdx.x;
  if (e >= (long)E2) return;
  int s, d; edge_sd(ei, e, s, d);
  float v = as2[s] + ad2[d];
  v = (v > 0.f) ? v : NEG_SLOPE * v;
  float ex = expf(v - dec_ordered(m2[d]));
  ex2[e] = ex;
  atomicAdd(&denom2[d], ex);
}

__global__ void edge_agg2_kernel(const long long* __restrict__ ei,
                                 const float* __restrict__ h2,
                                 const float* __restrict__ ex2,
                                 const float* __restrict__ denom2,
                                 float* __restrict__ agg2) {
  long g = (long)blockIdx.x * blockDim.x + threadIdx.x;
  long e = g >> 5; int lane = (int)(g & 31);
  if (e >= (long)E2 || lane >= NCLS) return;
  int s, d; edge_sd(ei, e, s, d);
  float coef = ex2[e] / denom2[d];
  atomicAdd(&agg2[(size_t)d * NCLS + lane], h2[(size_t)s * NCLS + lane] * coef);
}

__global__ void logsoftmax_kernel(const float* __restrict__ agg2, const float* __restrict__ b2,
                                  float* __restrict__ out) {
  int n = blockIdx.x * blockDim.x + threadIdx.x;
  if (n >= NNODE) return;
  float v[NCLS]; float mx = -INFINITY;
  #pragma unroll
  for (int c = 0; c < NCLS; ++c) { v[c] = agg2[(size_t)n * NCLS + c] + b2[c]; mx = fmaxf(mx, v[c]); }
  float sum = 0.f;
  #pragma unroll
  for (int c = 0; c < NCLS; ++c) sum += expf(v[c] - mx);
  float lse = mx + logf(sum);
  #pragma unroll
  for (int c = 0; c < NCLS; ++c) out[(size_t)n * NCLS + c] = v[c] - lse;
}

// ---------------- host ----------------
static inline int gridFor(long threads, int block) { return (int)((threads + block - 1) / block); }

extern "C" void kernel_launch(void* const* d_in, const int* in_sizes, int n_in,
                              void* d_out, int out_size, void* d_ws, size_t ws_size,
                              hipStream_t stream) {
  (void)in_sizes; (void)n_in; (void)out_size; (void)ws_size;
  const float*     x        = (const float*)d_in[0];
  const long long* ei       = (const long long*)d_in[1];   // int64 [2,E]
  const float*     W0       = (const float*)d_in[2];
  const float*     b0       = (const float*)d_in[3];
  const float*     W1       = (const float*)d_in[4];
  const float*     att_src1 = (const float*)d_in[5];
  const float*     att_dst1 = (const float*)d_in[6];
  const float*     b1       = (const float*)d_in[7];
  const float*     W2       = (const float*)d_in[8];
  const float*     att_src2 = (const float*)d_in[9];
  const float*     att_dst2 = (const float*)d_in[10];
  const float*     b2       = (const float*)d_in[11];
  float* out = (float*)d_out;

  char* w = (char*)d_ws; size_t off = 0;
  auto alloc = [&](size_t bytes) -> void* {
    off = (off + 255) & ~(size_t)255; void* p = w + off; off += bytes; return p;
  };
  __bf16* xb    = (__bf16*)alloc((size_t)NNODE * FIN * 2);
  __bf16* w0t   = (__bf16*)alloc((size_t)HIDD * FIN * 2);          // [64][128]
  __bf16* h0b   = (__bf16*)alloc((size_t)NNODE * HIDD * 2);
  __bf16* w1t   = (__bf16*)alloc((size_t)(HEADS*HIDD) * HIDD * 2); // [128][64]
  __bf16* w2tp  = (__bf16*)alloc((size_t)32 * FIN * 2);            // [32][128], rows>=18 zero
  __bf16* h1ab  = (__bf16*)alloc((size_t)NNODE * (HEADS*HIDD) * 2);
  float*  h1    = (float*)alloc((size_t)NNODE * (HEADS*HIDD) * 4);
  float*  as1   = (float*)alloc((size_t)NNODE * HEADS * 4);
  float*  ad1   = (float*)alloc((size_t)NNODE * HEADS * 4);
  float*  ex1   = (float*)alloc((size_t)E2 * HEADS * 4);
  float*  h2    = (float*)alloc((size_t)NNODE * NCLS * 4);
  float*  as2   = (float*)alloc((size_t)NNODE * 4);
  float*  ad2   = (float*)alloc((size_t)NNODE * 4);
  float*  ex2   = (float*)alloc((size_t)E2 * 4);
  // zero-init block (one memset): maxes (encoded), denominators, accumulators
  size_t zwords = (size_t)NNODE * (2 + 2 + 128 + 1 + 1 + 18);
  unsigned* zbase  = (unsigned*)alloc(zwords * 4);
  unsigned* m1     = zbase;
  float*    denom1 = (float*)(zbase + 2 * NNODE);
  float*    agg1   = denom1 + 2 * NNODE;
  unsigned* m2     = (unsigned*)(agg1 + (size_t)128 * NNODE);
  float*    denom2 = (float*)(m2 + NNODE);
  float*    agg2   = denom2 + NNODE;

  hipMemsetAsync(zbase, 0, zwords * 4, stream);

  // prep: bf16 conversions / weight transposes
  cvt_bf16_kernel<<<gridFor((long)NNODE * FIN, 256), 256, 0, stream>>>(x, xb, (long)NNODE * FIN);
  transpose_bf16_kernel<<<gridFor(64 * 128, 256), 256, 0, stream>>>(W0, w0t, FIN, HIDD, HIDD);
  transpose_bf16_kernel<<<gridFor(128 * 64, 256), 256, 0, stream>>>(W1, w1t, HIDD, HEADS*HIDD, HEADS*HIDD);
  transpose_bf16_kernel<<<gridFor(32 * 128, 256), 256, 0, stream>>>(W2, w2tp, FIN, NCLS, 32);

  const int rowTiles = NNODE / 16;   // 6250
  // h0 = relu(x@W0 + b0) -> bf16
  wmma_gemm_bf16_kernel<<<rowTiles, 32 * (HIDD/16), 0, stream>>>(
      xb, w0t, nullptr, h0b, b0, FIN, HIDD, HIDD, 1);
  // h1 = h0@W1 -> f32
  wmma_gemm_bf16_kernel<<<rowTiles, 32 * ((HEADS*HIDD)/16), 0, stream>>>(
      h0b, w1t, h1, nullptr, nullptr, HIDD, HEADS*HIDD, HEADS*HIDD, 0);

  attn_scores1_kernel<<<gridFor((long)NNODE * HEADS, 256), 256, 0, stream>>>(
      h1, att_src1, att_dst1, as1, ad1);

  edge_max1_kernel<<<gridFor((long)E2 * HEADS, 256), 256, 0, stream>>>(ei, as1, ad1, m1);
  edge_exp1_kernel<<<gridFor((long)E2 * HEADS, 256), 256, 0, stream>>>(ei, as1, ad1, m1, ex1, denom1);
  edge_agg1_kernel<<<gridFor((long)E2 * 32, 256), 256, 0, stream>>>(ei, h1, ex1, denom1, agg1);
  node_elu1_kernel<<<gridFor((long)NNODE * (HEADS*HIDD), 256), 256, 0, stream>>>(agg1, b1, h1ab);

  // h2 = elu_out @ W2 -> f32 (cols padded to 32, store 18)
  wmma_gemm_bf16_kernel<<<rowTiles, 32 * (32/16), 0, stream>>>(
      h1ab, w2tp, h2, nullptr, nullptr, FIN, 32, NCLS, 0);

  attn_scores2_kernel<<<gridFor(NNODE, 256), 256, 0, stream>>>(h2, att_src2, att_dst2, as2, ad2);
  edge_max2_kernel<<<gridFor((long)E2, 256), 256, 0, stream>>>(ei, as2, ad2, m2);
  edge_exp2_kernel<<<gridFor((long)E2, 256), 256, 0, stream>>>(ei, as2, ad2, m2, ex2, denom2);
  edge_agg2_kernel<<<gridFor((long)E2 * 32, 256), 256, 0, stream>>>(ei, h2, ex2, denom2, agg2);

  logsoftmax_kernel<<<gridFor(NNODE, 256), 256, 0, stream>>>(agg2, b2, out);
}
